// RefPhi4FlashDecoderLayer_56650618635020
// MI455X (gfx1250) — matrix operand order repaired and checked by reading, not compiled, not run on hardware
//
#include <hip/hip_runtime.h>
#include <hip/hip_bf16.h>
#include <math.h>

#define USE_TDM 1

typedef __attribute__((ext_vector_type(16))) __bf16 v16bf;
typedef __attribute__((ext_vector_type(8)))  float  v8f;
typedef __attribute__((ext_vector_type(4)))  unsigned int u32x4;
typedef __attribute__((ext_vector_type(8)))  int i32x8;
typedef __attribute__((ext_vector_type(4)))  int i32x4;

#define SEQ    2048
#define HID    2048
#define DIN    4096
#define DSTATE 16
#define DTRANK 128
#define DCONV  4
#define NHEADS 16
#define NKVH   4
#define HD     128
#define INTER  8192
#define XPROJ_N (DTRANK + 2*DSTATE)      /* 160  */
#define QKV_N   (NHEADS*HD + 2*NKVH*HD)  /* 3072 */
#define LAM_INIT 0.35550907f
#define EPSV     1e-5f

// LDS strides (ushorts): 40*2=80B rows for A (16B-aligned, conflict-free
// fragment reads), 72*2=144B rows for B, 24*2=48B groups for attention.
#define APAD  40
#define BPADG 72
#define BPAD  24

static __device__ __forceinline__ unsigned short f2bf(float f) {
  union { float f; unsigned u; } x; x.f = f;
  unsigned r = x.u + 0x7FFFu + ((x.u >> 16) & 1u);   // round-to-nearest-even
  return (unsigned short)(r >> 16);
}
static __device__ __forceinline__ v8f vz8() {
  v8f z;
#pragma unroll
  for (int i = 0; i < 8; ++i) z[i] = 0.f;
  return z;
}
static __device__ __forceinline__ float siluf(float v) { return v / (1.f + __expf(-v)); }

#if USE_TDM
// LDS byte offset of a __shared__ object (generic -> AS(3) -> integer)
static __device__ __forceinline__ unsigned lds_off(const void* p) {
  return (unsigned)(unsigned long long)
      (__attribute__((address_space(3))) const void*)p;
}
// Issue a TDM 2D tile load: global (strided, bf16) -> LDS (padded rows).
// dim0_rem/dim1_rem: remaining tensor extent from tile start (OOB -> zero).
// pad_iv: pad-interval code (3 = every 16 DWORDs, 4 = every 32 DWORDs);
// pad_am: pad-amount code (3 = 4 DWORDs = 16 bytes).
static __device__ __forceinline__ void tdm_load_2d(
    unsigned ldsaddr, const void* gaddr, int dim0_rem, int dim1_rem,
    int tile0, int tile1, long long stride0, int pad_iv, int pad_am)
{
  unsigned long long ga = (unsigned long long)(size_t)gaddr;
  u32x4 g0;
  g0[0] = 1u;                                   // count=1 (valid), user mode
  g0[1] = ldsaddr;                              // lds_addr
  g0[2] = (unsigned)(ga & 0xFFFFFFFFu);         // global_addr[31:0]
  g0[3] = (unsigned)((ga >> 32) & 0x01FFFFFFu)  // global_addr[56:32]
          | (2u << 30);                         // type=2 (image)
  i32x8 g1;
  g1[0] = (1 << 16)                             // data_size=1 (2 bytes)
        | (1 << 20)                             // pad_enable
        | (pad_iv << 22) | (pad_am << 25);
  g1[1] = (dim0_rem & 0xFFFF) << 16;            // tensor_dim0[15:0]
  g1[2] = (int)(((unsigned)dim0_rem >> 16) | ((unsigned)(dim1_rem & 0xFFFF) << 16));
  g1[3] = (int)(((unsigned)dim1_rem >> 16) | ((unsigned)tile0 << 16));  // tile_dim0
  g1[4] = tile1;                                // tile_dim1 (tile_dim2=0)
  g1[5] = (int)(stride0 & 0xFFFFFFFF);          // tensor_dim0_stride[31:0]
  g1[6] = (int)((stride0 >> 32) & 0xFFFF);      // stride[47:32]; dim1_stride=0
  g1[7] = 0;
  i32x4 z4 = {0, 0, 0, 0};
  i32x8 z8 = {0, 0, 0, 0, 0, 0, 0, 0};
  __builtin_amdgcn_tensor_load_to_lds(g0, g1, z4, z4, z8, 0);
}
#endif

// ---------------------------------------------------------------------------
// Weight transpose+convert: W[N][K] f32 -> Wt[K][N] bf16. N,K multiples of 32.
// ---------------------------------------------------------------------------
__launch_bounds__(256)
__global__ void wconv_kernel(const float* __restrict__ W, unsigned short* __restrict__ Wt,
                             int N, int K)
{
  __shared__ float t[32][33];
  const int k0 = blockIdx.x * 32, n0 = blockIdx.y * 32;
  const int c = threadIdx.x & 31, r8 = threadIdx.x >> 5;
#pragma unroll
  for (int s = 0; s < 4; ++s) {
    int r = r8 + s * 8;
    t[r][c] = W[(size_t)(n0 + r) * K + k0 + c];
  }
  __syncthreads();
#pragma unroll
  for (int s = 0; s < 4; ++s) {
    int r = r8 + s * 8;                   // r = k within tile
    Wt[(size_t)(k0 + r) * N + n0 + c] = f2bf(t[c][r]);
  }
}

// ---------------------------------------------------------------------------
// bf16 WMMA GEMM: C[M,N] = A[M,K] * Wt[K,N] (+bias) (+C) (+res)
// A: bf16 [M][lda]; Wt: bf16 [K][N] (pre-transposed). M%128==0, K%32==0.
// Tile 128x64, 8 waves, K-step 32, double-buffered LDS.
// Staging via Tensor Data Mover: wave 0 issues 2 DMA descriptors per step,
// synchronized with s_wait_tensorcnt + workgroup barrier.
// ---------------------------------------------------------------------------
__launch_bounds__(256)
__global__ void gemm_bf16_wmma(const unsigned short* __restrict__ A, int lda,
                               const unsigned short* __restrict__ Wt,
                               const float* __restrict__ bias,
                               const float* __restrict__ res,
                               float* __restrict__ C,
                               unsigned short* __restrict__ Cbf,
                               int ldc, int M, int N, int K, int addC)
{
  __shared__ alignas(16) unsigned short lA[2][128 * APAD];   // [row][k]
  __shared__ alignas(16) unsigned short lB[2][32 * BPADG];   // [k][n]
  const int tid  = threadIdx.x;
  const int wave = tid >> 5;
  const int lane = tid & 31;
  const int hh   = lane >> 4;
  const int ln   = lane & 15;
  const int m0   = blockIdx.y * 128;
  const int n0   = blockIdx.x * 64;
  const int kk   = ln + (hh << 4);   // B-operand K lane index

#if USE_TDM
  auto stage = [&](int k0, int buf) {
    // A tile: x = k (32), y = m (128), row stride lda, pad 16B per 64B row
    tdm_load_2d(lds_off(&lA[buf][0]),
                A + (size_t)m0 * lda + k0, K - k0, M - m0,
                32, 128, (long long)lda, 3, 3);
    // B tile: x = n (64), y = k (32), row stride N, pad 16B per 128B row
    tdm_load_2d(lds_off(&lB[buf][0]),
                Wt + (size_t)k0 * N + n0, N - n0, K - k0,
                64, 32, (long long)N, 4, 3);
  };
#else
  auto stage = [&](int k0, int buf) {
    unsigned short* dstA = lA[buf];
#pragma unroll
    for (int s = 0; s < 2; ++s) {
      int ch = tid + s * 256;
      int r = ch >> 2, c = (ch & 3) * 8;
      u32x4 v = *reinterpret_cast<const u32x4*>(A + (size_t)(m0 + r) * lda + k0 + c);
      *reinterpret_cast<u32x4*>(&dstA[r * APAD + c]) = v;
    }
    unsigned short* dstB = lB[buf];
    int r = tid >> 3, c = (tid & 7) * 8;
    u32x4 v = *reinterpret_cast<const u32x4*>(Wt + (size_t)(k0 + r) * N + n0 + c);
    *reinterpret_cast<u32x4*>(&dstB[r * BPADG + c]) = v;
  };
#endif

  v8f acc[4];
#pragma unroll
  for (int t = 0; t < 4; ++t) acc[t] = vz8();

  const int nk = K >> 5;
#if USE_TDM
  if (wave == 0) stage(0, 0);
#else
  stage(0, 0);
#endif
  for (int ki = 0; ki < nk; ++ki) {
    const int buf = ki & 1;
#if USE_TDM
    if (wave == 0) __builtin_amdgcn_s_wait_tensorcnt(0);
    __syncthreads();                   // publish tile ki to all waves
    if (ki + 1 < nk && wave == 0) stage((ki + 1) << 5, buf ^ 1);
#else
    __syncthreads();
    if (ki + 1 < nk) stage((ki + 1) << 5, buf ^ 1);
#endif
    const unsigned short* sA = lA[buf];
    const unsigned short* sB = lB[buf];
    union { v16bf v; unsigned short u[16]; } af;
    const int arow = (wave * 16 + ln) * APAD;
#pragma unroll
    for (int e = 0; e < 16; ++e) {
      int k = ((e >> 3) << 4) + (hh << 3) + (e & 7);
      af.u[e] = sA[arow + k];
    }
#pragma unroll
    for (int t = 0; t < 4; ++t) {
      union { v16bf v; unsigned short u[16]; } bf;
      const int bbase = t * 16 + ln;
#pragma unroll
      for (int e = 0; e < 16; ++e) bf.u[e] = sB[bbase + (((e >> 3) << 4) + (hh << 3) + (e & 7)) * BPADG];
      acc[t] = __builtin_amdgcn_wmma_f32_16x16x32_bf16(
          false, af.v, false, bf.v, (short)0, acc[t], false, false);
    }
  }

  const bool full = (n0 + 64 <= N);
#pragma unroll
  for (int t = 0; t < 4; ++t) {
#pragma unroll
    for (int e = 0; e < 8; ++e) {
      int row = m0 + wave * 16 + e + 8 * hh;
      int col = n0 + t * 16 + ln;
      if (full || col < N) {
        float v = acc[t][e];
        if (bias) v += bias[col];
        size_t idx = (size_t)row * ldc + col;
        if (Cbf) {
          Cbf[idx] = f2bf(v);
        } else {
          if (res)  v += res[idx];
          if (addC) v += C[idx];
          C[idx] = v;
        }
      }
    }
  }
}

// ---------------------------------------------------------------------------
// Differential flash attention on bf16 qkv: one wave per (16-query tile,
// head, q-half). P = softmax(q_h k_h^T) once; P@v1 and P@v2 together.
// ---------------------------------------------------------------------------
__launch_bounds__(32)
__global__ void flash_attn_diff(const unsigned short* __restrict__ qkv,
                                float* __restrict__ attn1,
                                float* __restrict__ attn2)
{
  __shared__ alignas(16) unsigned short lK [8 * 32 * BPAD];
  __shared__ alignas(16) unsigned short lVa[8 * 32 * BPAD];
  __shared__ alignas(16) unsigned short lVb[8 * 32 * BPAD];
  __shared__ alignas(16) unsigned short lP [16 * APAD];

  const int lane = threadIdx.x & 31;
  const int hh   = lane >> 4;
  const int ln   = lane & 15;
  const int kkl  = ln + (hh << 4);
  const int q0   = blockIdx.x * 16;
  const int head = blockIdx.y;        // 0..7 (post-split head)
  const int qh   = blockIdx.z;        // 0 -> (q1,k1), 1 -> (q2,k2)
  const int qoff  = (2 * head + qh) * HD;
  const int koff  = NHEADS * HD + (2 * (head >> 2) + qh) * HD;
  const int voffA = NHEADS * HD + NKVH * HD + (2 * (head >> 2)) * HD; // v1
  const int voffB = voffA + HD;                                      // v2
  const float scale = 0.08838834764831845f;   // 1/sqrt(128)

  union F16 { v16bf v; unsigned short u[16]; u32x4 q[2]; };
  F16 qf[4];
#pragma unroll
  for (int c = 0; c < 4; ++c) {
    const unsigned short* qp =
        qkv + (size_t)(q0 + ln) * QKV_N + qoff + c * 32 + (hh << 3);
    qf[c].q[0] = *reinterpret_cast<const u32x4*>(qp);        // e0..7
    qf[c].q[1] = *reinterpret_cast<const u32x4*>(qp + 16);   // e8..15
  }

  v8f acca[8], accb[8];
#pragma unroll
  for (int j = 0; j < 8; ++j) { acca[j] = vz8(); accb[j] = vz8(); }
  float mrow[8], lrow[8];
#pragma unroll
  for (int e = 0; e < 8; ++e) { mrow[e] = -3.0e38f; lrow[e] = 0.f; }

  for (int kb = 0; kb <= q0; kb += 32) {
    // stage 32 keys x 128 hd of K, V1, V2 (bf16): 8 elems per lane-chunk
    for (int s = 0; s < 16; ++s) {
      int ch = lane + s * 32;
      int r = ch >> 4, cc = (ch & 15) * 8;   // r = key, cc = hd (mult of 8)
      const unsigned short* base = qkv + (size_t)(kb + r) * QKV_N;
      union { u32x4 q; unsigned short u[8]; } kv;
      kv.q = *reinterpret_cast<const u32x4*>(base + koff + cc);
      // K: elem = key-in-tile, lane-dim = hd (stride BPAD)
      int kbse = (((r >> 4) * 4 + (cc >> 5)) * 32 + (cc & 31)) * BPAD + (r & 15);
#pragma unroll
      for (int j = 0; j < 8; ++j) lK[kbse + j * BPAD] = kv.u[j];
      // V: elem = hd-in-tile (contiguous) -> single b128 store
      int vbse = ((cc >> 4) * 32 + r) * BPAD + (cc & 15);
      *reinterpret_cast<u32x4*>(&lVa[vbse]) =
          *reinterpret_cast<const u32x4*>(base + voffA + cc);
      *reinterpret_cast<u32x4*>(&lVb[vbse]) =
          *reinterpret_cast<const u32x4*>(base + voffB + cc);
    }
    __syncthreads();

    // scores: two 16x16 tiles over 32 keys, contraction over HD=128
    v8f sc[2]; sc[0] = vz8(); sc[1] = vz8();
#pragma unroll
    for (int t = 0; t < 2; ++t) {
#pragma unroll
      for (int c = 0; c < 4; ++c) {
        F16 bf;
        const int bbase = (t * 4 + c) * 32 * BPAD + ln;
#pragma unroll
        for (int e = 0; e < 16; ++e) bf.u[e] = lK[bbase + (((e >> 3) << 4) + (hh << 3) + (e & 7)) * BPAD];
        sc[t] = __builtin_amdgcn_wmma_f32_16x16x32_bf16(
            false, qf[c].v, false, bf.v, (short)0, sc[t], false, false);
      }
    }

    float scf[2][8];
#pragma unroll
    for (int t = 0; t < 2; ++t)
#pragma unroll
      for (int e = 0; e < 8; ++e) {
        int qpos = q0 + e + 8 * hh;
        int kpos = kb + t * 16 + ln;
        float s = sc[t][e] * scale;
        scf[t][e] = (kpos > qpos) ? -3.0e38f : s;
      }

    // online softmax; a row of the C layout lives in one 16-lane half
    float fac[8];
#pragma unroll
    for (int e = 0; e < 8; ++e) {
      float r0 = scf[0][e], r1 = scf[1][e];
#pragma unroll
      for (int m = 8; m >= 1; m >>= 1) {
        r0 = fmaxf(r0, __shfl_xor(r0, m, 32));
        r1 = fmaxf(r1, __shfl_xor(r1, m, 32));
      }
      float mnew = fmaxf(mrow[e], fmaxf(r0, r1));
      fac[e] = __expf(mrow[e] - mnew);
      mrow[e] = mnew;
    }
    float p[2][8];
#pragma unroll
    for (int t = 0; t < 2; ++t)
#pragma unroll
      for (int e = 0; e < 8; ++e) p[t][e] = __expf(scf[t][e] - mrow[e]);
#pragma unroll
    for (int e = 0; e < 8; ++e) {
      float s0 = p[0][e] + p[1][e];
#pragma unroll
      for (int m = 8; m >= 1; m >>= 1) s0 += __shfl_xor(s0, m, 32);
      lrow[e] = lrow[e] * fac[e] + s0;
    }
#pragma unroll
    for (int j = 0; j < 8; ++j)
#pragma unroll
      for (int e = 0; e < 8; ++e) { acca[j][e] *= fac[e]; accb[j][e] *= fac[e]; }

    // P: C-layout -> LDS (padded rows) -> A-layout (two contiguous runs/lane)
#pragma unroll
    for (int t = 0; t < 2; ++t)
#pragma unroll
      for (int e = 0; e < 8; ++e)
        lP[(e + 8 * hh) * APAD + t * 16 + ln] = f2bf(p[t][e]);
    __syncthreads();

    F16 pf;
#pragma unroll
    for (int e = 0; e < 16; ++e) {
      int k = ((e >> 3) << 4) + (hh << 3) + (e & 7);
      pf.u[e] = lP[ln * APAD + k];
    }
#pragma unroll
    for (int j = 0; j < 8; ++j) {
      F16 ba, bb;
      const int vbase = j * 32 * BPAD + ln;
#pragma unroll
      for (int e = 0; e < 16; ++e) {
        int k = ((e >> 3) << 4) + (hh << 3) + (e & 7);
        ba.u[e] = lVa[vbase + k * BPAD];
        bb.u[e] = lVb[vbase + k * BPAD];
      }
      acca[j] = __builtin_amdgcn_wmma_f32_16x16x32_bf16(
          false, pf.v, false, ba.v, (short)0, acca[j], false, false);
      accb[j] = __builtin_amdgcn_wmma_f32_16x16x32_bf16(
          false, pf.v, false, bb.v, (short)0, accb[j], false, false);
    }
    __syncthreads();
  }

  float* outb = (qh == 0) ? attn1 : attn2;   // attn1=[a11|a12], attn2=[a21|a22]
#pragma unroll
  for (int e = 0; e < 8; ++e) {
    int row = q0 + e + 8 * hh;
    float rl = 1.f / lrow[e];
#pragma unroll
    for (int j = 0; j < 8; ++j) {
      int col = j * 16 + ln;
      outb[(size_t)row * 2048 + head * 256 +       col] = acca[j][e] * rl;
      outb[(size_t)row * 2048 + head * 256 + 128 + col] = accb[j][e] * rl;
    }
  }
}

// ---------------------------------------------------------------------------
// Light kernels
// ---------------------------------------------------------------------------
__launch_bounds__(256)
__global__ void rmsnorm_kernel(const float* __restrict__ x, const float* __restrict__ w,
                               unsigned short* __restrict__ out, int H)
{
  __shared__ float red[256];
  int row = blockIdx.x, tid = threadIdx.x;
  const float* xr = x + (size_t)row * H;
  float s = 0.f;
  for (int c = tid; c < H; c += 256) { float v = xr[c]; s += v * v; }
  red[tid] = s; __syncthreads();
  for (int o = 128; o > 0; o >>= 1) { if (tid < o) red[tid] += red[tid + o]; __syncthreads(); }
  float inv = rsqrtf(red[0] / H + EPSV);
  unsigned short* orow = out + (size_t)row * H;
  for (int c = tid; c < H; c += 256) orow[c] = f2bf(w[c] * xr[c] * inv);
}

__global__ void conv_silu_kernel(const float* __restrict__ xz, const float* __restrict__ cw,
                                 const float* __restrict__ cb,
                                 float* __restrict__ outf, unsigned short* __restrict__ outb)
{
  int idx = blockIdx.x * 256 + threadIdx.x;
  if (idx >= SEQ * DIN) return;
  int t = idx / DIN, d = idx % DIN;
  float acc = cb[d];
#pragma unroll
  for (int i = 0; i < DCONV; ++i) {
    int tt = t - (DCONV - 1) + i;
    if (tt >= 0) acc += xz[(size_t)tt * 2 * DIN + d] * cw[d * DCONV + i];
  }
  float v = siluf(acc);
  outf[idx] = v;
  outb[idx] = f2bf(v);
}

__global__ void dtin_kernel(const float* __restrict__ dbc, unsigned short* __restrict__ o)
{
  int i = blockIdx.x * 256 + threadIdx.x;
  if (i >= SEQ * DTRANK) return;
  int s = i >> 7, c = i & 127;
  o[i] = f2bf(dbc[(size_t)s * XPROJ_N + c]);
}

__global__ void softplus_kernel(float* __restrict__ p, int n)
{
  int i = blockIdx.x * 256 + threadIdx.x;
  if (i >= n) return;
  float v = p[i];
  p[i] = (v > 20.f) ? v : log1pf(__expf(v));
}

__launch_bounds__(256)
__global__ void scan_kernel(const float* __restrict__ xs, const float* __restrict__ dt,
                            const float* __restrict__ dbc, const float* __restrict__ A_log,
                            const float* __restrict__ Dp, float* __restrict__ y)
{
  int d = blockIdx.x * 256 + threadIdx.x;   // channel 0..DIN-1
  float Ar[DSTATE];
#pragma unroll
  for (int n = 0; n < DSTATE; ++n) Ar[n] = -__expf(A_log[(size_t)d * DSTATE + n]);
  float Dv = Dp[d];
  float st[DSTATE];
#pragma unroll
  for (int n = 0; n < DSTATE; ++n) st[n] = 0.f;
  for (int t = 0; t < SEQ; ++t) {
    float dtv = dt[(size_t)t * DIN + d];
    float xv  = xs[(size_t)t * DIN + d];
    const float* Bp = dbc + (size_t)t * XPROJ_N + DTRANK;
    const float* Cp = Bp + DSTATE;
    float xdt = xv * dtv, acc = 0.f;
#pragma unroll
    for (int n = 0; n < DSTATE; ++n) {
      float dA = __expf(dtv * Ar[n]);
      st[n] = st[n] * dA + xdt * Bp[n];
      acc += st[n] * Cp[n];
    }
    y[(size_t)t * DIN + d] = acc + Dv * xv;
  }
}

__global__ void ygate_kernel(const float* __restrict__ y, const float* __restrict__ xz,
                             unsigned short* __restrict__ o)
{
  int idx = blockIdx.x * 256 + threadIdx.x;
  if (idx >= SEQ * DIN) return;
  int t = idx / DIN, d = idx % DIN;
  float z = xz[(size_t)t * 2 * DIN + DIN + d];
  o[idx] = f2bf(y[idx] * siluf(z));
}

__launch_bounds__(128)
__global__ void lambda_kernel(const float* lq1, const float* lk1,
                              const float* lq2, const float* lk2, float* lam)
{
  __shared__ float r1[128], r2[128];
  int i = threadIdx.x;
  r1[i] = lq1[i] * lk1[i];
  r2[i] = lq2[i] * lk2[i];
  __syncthreads();
  for (int o = 64; o > 0; o >>= 1) {
    if (i < o) { r1[i] += r1[i + o]; r2[i] += r2[i + o]; }
    __syncthreads();
  }
  if (i == 0) lam[0] = __expf(r1[0]) - __expf(r2[0]) + LAM_INIT;
}

__launch_bounds__(256)
__global__ void diff_combine_kernel(const float* __restrict__ a1, const float* __restrict__ a2,
                                    const float* __restrict__ lamp,
                                    const float* __restrict__ subw,
                                    unsigned short* __restrict__ out)
{
  __shared__ float red[256];
  int s   = blockIdx.x >> 3;
  int hd8 = blockIdx.x & 7;
  int c   = threadIdx.x;
  float lam = lamp[0];
  size_t idx = (size_t)s * 2048 + hd8 * 256 + c;
  float t = a1[idx] - lam * a2[idx];
  red[c] = t * t; __syncthreads();
  for (int o = 128; o > 0; o >>= 1) { if (c < o) red[c] += red[c + o]; __syncthreads(); }
  float inv = rsqrtf(red[0] / 256.f + EPSV);
  out[idx] = f2bf(subw[c] * t * inv * (1.f - LAM_INIT));
}

__global__ void glu_kernel(const float* __restrict__ gu, unsigned short* __restrict__ act)
{
  int idx = blockIdx.x * 256 + threadIdx.x;
  if (idx >= SEQ * INTER) return;
  int t = idx / INTER, j = idx % INTER;
  float g = gu[(size_t)t * 2 * INTER + j];
  float u = gu[(size_t)t * 2 * INTER + INTER + j];
  act[idx] = f2bf(u * siluf(g));
}

// ---------------------------------------------------------------------------
extern "C" void kernel_launch(void* const* d_in, const int* in_sizes, int n_in,
                              void* d_out, int out_size, void* d_ws, size_t ws_size,
                              hipStream_t stream)
{
  (void)in_sizes; (void)n_in; (void)out_size; (void)ws_size;
  const float* x           = (const float*)d_in[0];
  const float* mamba_norm_w= (const float*)d_in[1];
  const float* in_proj_w   = (const float*)d_in[2];
  const float* conv_w      = (const float*)d_in[3];
  const float* conv_b      = (const float*)d_in[4];
  const float* x_proj_w    = (const float*)d_in[5];
  const float* dt_proj_w   = (const float*)d_in[6];
  const float* dt_proj_b   = (const float*)d_in[7];
  const float* A_log       = (const float*)d_in[8];
  const float* D_param     = (const float*)d_in[9];
  const float* mamba_out_w = (const float*)d_in[10];
  const float* attn_norm_w = (const float*)d_in[11];
  const float* wqkv_w      = (const float*)d_in[12];
  const float* wqkv_b      = (const float*)d_in[13];
  const float* out_proj_w  = (const float*)d_in[14];
  const float* out_proj_b  = (const float*)d_in[15];
  const float* lambda_q1   = (const float*)d_in[16];
  const float* lambda_k1   = (const float*)d_in[17];
  const float* lambda_q2   = (const float*)d_in[18];
  const float* lambda_k2   = (const float*)d_in[19];
  const float* subln_w     = (const float*)d_in[20];
  const float* mlp_norm_w  = (const float*)d_in[21];
  const float* fc1_w       = (const float*)d_in[22];
  const float* fc2_w       = (const float*)d_in[23];
  float* h = (float*)d_out;   // residual stream lives in d_out

  typedef unsigned short us;
  size_t off = 0;
  auto walloc = [&](size_t bytes) -> void* {
    void* p = (char*)d_ws + off;
    off += (bytes + 511) & ~(size_t)511;
    return p;
  };
  // bf16 activation buffers
  us* hn    = (us*)walloc((size_t)SEQ * HID * 2);
  us* xcb   = (us*)walloc((size_t)SEQ * DIN * 2);
  us* dtin  = (us*)walloc((size_t)SEQ * DTRANK * 2);
  us* ybbf  = (us*)walloc((size_t)SEQ * DIN * 2);
  us* qkvb  = (us*)walloc((size_t)SEQ * QKV_N * 2);
  us* aobf  = (us*)walloc((size_t)SEQ * 2048 * 2);
  us* actbf = (us*)walloc((size_t)SEQ * INTER * 2);
  // f32 buffers
  float* xz   = (float*)walloc((size_t)SEQ * 2 * DIN * 4);
  float* xc   = (float*)walloc((size_t)SEQ * DIN * 4);
  float* dbc  = (float*)walloc((size_t)SEQ * XPROJ_N * 4);
  float* dtb  = (float*)walloc((size_t)SEQ * DIN * 4);
  float* yb   = (float*)walloc((size_t)SEQ * DIN * 4);
  float* a1   = (float*)walloc((size_t)SEQ * 2048 * 4);
  float* a2   = (float*)walloc((size_t)SEQ * 2048 * 4);
  float* lamp = (float*)walloc(256);
  float* gu   = (float*)walloc((size_t)SEQ * 2 * INTER * 4);
  // transposed bf16 weights (one conversion pass per launch)
  us* wt_in   = (us*)walloc((size_t)HID    * 2 * DIN * 2);
  us* wt_xp   = (us*)walloc((size_t)DIN    * XPROJ_N * 2 + 256);
  us* wt_dt   = (us*)walloc((size_t)DTRANK * DIN * 2);
  us* wt_mo   = (us*)walloc((size_t)DIN    * HID * 2);
  us* wt_qkv  = (us*)walloc((size_t)HID    * QKV_N * 2);
  us* wt_op   = (us*)walloc((size_t)HID    * HID * 2);
  us* wt_fc1  = (us*)walloc((size_t)HID    * 2 * INTER * 2);
  us* wt_fc2  = (us*)walloc((size_t)INTER  * HID * 2);

  auto wconv = [&](const float* W, us* Wt, int N, int K) {
    dim3 g(K / 32, N / 32);
    wconv_kernel<<<g, 256, 0, stream>>>(W, Wt, N, K);
  };
  auto gemm = [&](const us* A, int lda, const us* Wt, const float* bias,
                  const float* res, float* C, us* Cbf, int ldc,
                  int M, int N, int K, int addC) {
    dim3 g((N + 63) / 64, (M + 127) / 128);
    gemm_bf16_wmma<<<g, 256, 0, stream>>>(A, lda, Wt, bias, res, C, Cbf, ldc,
                                          M, N, K, addC);
  };

  // ---- one-time weight transpose/convert (per launch) ----
  wconv(in_proj_w,   wt_in,  2 * DIN,  HID);
  wconv(x_proj_w,    wt_xp,  XPROJ_N,  DIN);
  wconv(dt_proj_w,   wt_dt,  DIN,      DTRANK);
  wconv(mamba_out_w, wt_mo,  HID,      DIN);
  wconv(wqkv_w,      wt_qkv, QKV_N,    HID);
  wconv(out_proj_w,  wt_op,  HID,      HID);
  wconv(fc1_w,       wt_fc1, 2 * INTER, HID);
  wconv(fc2_w,       wt_fc2, HID,      INTER);

  // ---- Mamba mixer ----
  rmsnorm_kernel<<<SEQ, 256, 0, stream>>>(x, mamba_norm_w, hn, HID);
  gemm(hn, HID, wt_in, nullptr, nullptr, xz, nullptr, 2 * DIN, SEQ, 2 * DIN, HID, 0);
  conv_silu_kernel<<<(SEQ * DIN + 255) / 256, 256, 0, stream>>>(xz, conv_w, conv_b, xc, xcb);
  gemm(xcb, DIN, wt_xp, nullptr, nullptr, dbc, nullptr, XPROJ_N, SEQ, XPROJ_N, DIN, 0);
  dtin_kernel<<<(SEQ * DTRANK + 255) / 256, 256, 0, stream>>>(dbc, dtin);
  gemm(dtin, DTRANK, wt_dt, dt_proj_b, nullptr, dtb, nullptr, DIN, SEQ, DIN, DTRANK, 0);
  softplus_kernel<<<(SEQ * DIN + 255) / 256, 256, 0, stream>>>(dtb, SEQ * DIN);
  scan_kernel<<<DIN / 256, 256, 0, stream>>>(xc, dtb, dbc, A_log, D_param, yb);
  ygate_kernel<<<(SEQ * DIN + 255) / 256, 256, 0, stream>>>(yb, xz, ybbf);
  gemm(ybbf, DIN, wt_mo, nullptr, x, h, nullptr, HID, SEQ, HID, DIN, 0); // h = x + mamba

  // ---- Differential attention ----
  rmsnorm_kernel<<<SEQ, 256, 0, stream>>>(h, attn_norm_w, hn, HID);
  gemm(hn, HID, wt_qkv, wqkv_b, nullptr, nullptr, qkvb, QKV_N, SEQ, QKV_N, HID, 0);
  lambda_kernel<<<1, 128, 0, stream>>>(lambda_q1, lambda_k1, lambda_q2, lambda_k2, lamp);
  {
    dim3 g(SEQ / 16, NHEADS / 2, 2);
    flash_attn_diff<<<g, 32, 0, stream>>>(qkvb, a1, a2);
  }
  diff_combine_kernel<<<SEQ * 8, 256, 0, stream>>>(a1, a2, lamp, subln_w, aobf);
  gemm(aobf, HID, wt_op, out_proj_b, nullptr, h, nullptr, HID, SEQ, HID, HID, 1); // h += attn

  // ---- MLP ----
  rmsnorm_kernel<<<SEQ, 256, 0, stream>>>(h, mlp_norm_w, hn, HID);
  gemm(hn, HID, wt_fc1, nullptr, nullptr, gu, nullptr, 2 * INTER, SEQ, 2 * INTER, HID, 0);
  glu_kernel<<<(SEQ * INTER + 255) / 256, 256, 0, stream>>>(gu, actbf);
  gemm(actbf, INTER, wt_fc2, nullptr, nullptr, h, nullptr, HID, SEQ, HID, INTER, 1); // h += mlp
}
